// TemporalAttention_72181220376900
// MI455X (gfx1250) — matrix operand order repaired and checked by reading, not compiled, not run on hardware
//
#include <hip/hip_runtime.h>
#include <math.h>

// MI455X / gfx1250: wave32, WMMA f16 16x16x32 w/ f32 accumulate, TDM staging.
typedef __attribute__((ext_vector_type(16))) _Float16 v16h;
typedef __attribute__((ext_vector_type(8)))  _Float16 v8h;
typedef __attribute__((ext_vector_type(8)))  float    v8f;
typedef unsigned int v4u __attribute__((ext_vector_type(4)));
typedef int          v8i __attribute__((ext_vector_type(8)));
typedef int          v4i __attribute__((ext_vector_type(4)));

#define D_MODEL 256
#define NB      8
#define NTOK    4096            // tokens per batch (16*256)
#define ROWS    (NB * NTOK)     // 32768 total rows

// padded LDS row strides (halves) -> conflict-free b128 access
#define K_LDS_STRIDE 264        // 256 + 8 pad (4 dwords per 128-dword row)
#define V_LDS_STRIDE 40         // 32 + 8 pad  (4 dwords per 16-dword row)

#if defined(__has_builtin)
#if __has_builtin(__builtin_amdgcn_tensor_load_to_lds)
#define USE_TDM 1
#endif
#endif
#ifndef USE_TDM
#define USE_TDM 0
#endif

static __device__ __forceinline__ v8f wmma_f16(v16h a, v16h b, v8f c) {
  return __builtin_amdgcn_wmma_f32_16x16x32_f16(false, a, false, b, (short)0, c,
                                                false, false);
}

// ---- cross-lane helpers (wave32) -----------------------------------------
static __device__ __forceinline__ float lane_bcast(float v, int srcLane) {
  return __int_as_float(
      __builtin_amdgcn_ds_bpermute(srcLane << 2, __float_as_int(v)));
}
static __device__ __forceinline__ float xor16_max(float v) {
  float o = __int_as_float(__builtin_amdgcn_ds_bpermute(
      ((int)(threadIdx.x ^ 16) & 31) << 2, __float_as_int(v)));
  return fmaxf(v, o);
}
static __device__ __forceinline__ float xor16_add(float v) {
  float o = __int_as_float(__builtin_amdgcn_ds_bpermute(
      ((int)(threadIdx.x ^ 16) & 31) << 2, __float_as_int(v)));
  return v + o;
}

// ---- fragment loaders -----------------------------------------------------
// A operand (16x32 f16): lane L: row m=L%16; halves h<8 -> k=k0+g*8+h,
//                        h>=8 -> k=k0+16+g*8+(h-8).  p = rowbase + k0 + g*8
static __device__ __forceinline__ v16h load_fragA_f16(const _Float16* p) {
  v8h lo = *(const v8h*)p;
  v8h hi = *(const v8h*)(p + 16);
  v16h r;
#pragma unroll
  for (int i = 0; i < 8; ++i) { r[i] = lo[i]; r[8 + i] = hi[i]; }
  return r;
}
// B operand (32x16 f16): lane L: col n=L%16; halves h<8 -> k=k0+g*8+h,
//                        h>=8 -> k=k0+16+g*8+(h-8).  p = colbase + k0 + g*8
static __device__ __forceinline__ v16h load_fragB_f16(const _Float16* p) {
  v8h lo = *(const v8h*)p;
  v8h hi = *(const v8h*)(p + 16);
  v16h r;
#pragma unroll
  for (int i = 0; i < 8; ++i) { r[i] = lo[i]; r[8 + i] = hi[i]; }
  return r;
}
static __device__ __forceinline__ v16h cvt_fragA_f32(const float* p) {
  v8f lo = *(const v8f*)p;
  v8f hi = *(const v8f*)(p + 16);
  v16h r;
#pragma unroll
  for (int i = 0; i < 8; ++i) { r[i] = (_Float16)lo[i]; r[8 + i] = (_Float16)hi[i]; }
  return r;
}
static __device__ __forceinline__ v16h cvt_fragB_f32(const float* p) {
  v8f lo = *(const v8f*)p;
  v8f hi = *(const v8f*)(p + 16);
  v16h r;
#pragma unroll
  for (int i = 0; i < 8; ++i) { r[i] = (_Float16)lo[i]; r[8 + i] = (_Float16)hi[i]; }
  return r;
}

#if USE_TDM
// One TENSOR_LOAD_TO_LDS moving a 2-D tile (tile_d1 rows x tile_d0 halves)
// from global (row stride = stride0 halves) into LDS with hardware row
// padding (pad after 2^(pad_i+1) dwords, pad_a+1 dwords inserted).
// This toolchain's builtin is the 6-arg form:
//   (uint32x4 g0, int32x8 g1, int32x4 g2, int32x4 g3, int32x8 extra, i32 cpol)
static __device__ __forceinline__ void tdm_load_tile_f16(
    const _Float16* gaddr, unsigned lds_off, unsigned tensor_d0,
    unsigned tensor_d1, unsigned stride0, unsigned tile_d0, unsigned tile_d1,
    unsigned pad_i, unsigned pad_a) {
  unsigned long long ga = (unsigned long long)(uintptr_t)gaddr;
  v4u g0;
  g0[0] = 1u;                                   // count=1, user descriptor
  g0[1] = lds_off;                              // LDS byte address
  g0[2] = (unsigned)ga;                         // global_addr[31:0]
  g0[3] = (unsigned)((ga >> 32) & 0x1FFFFFFu)   // global_addr[56:32]
          | 0x80000000u;                        // type=2 ("image")
  v8i g1;
  g1[0] = (int)(0x00010000u                     // data_size=1 -> 2 bytes
                | (1u << 20)                    // pad_enable
                | (pad_i << 22) | (pad_a << 25));
  g1[1] = (int)((tensor_d0 & 0xFFFFu) << 16);   // tensor_dim0[15:0]
  g1[2] = (int)((tensor_d0 >> 16) | ((tensor_d1 & 0xFFFFu) << 16));
  g1[3] = (int)((tensor_d1 >> 16) | (tile_d0 << 16));
  g1[4] = (int)tile_d1;                         // tile_dim1 (tile_dim2=0)
  g1[5] = (int)stride0;                         // tensor_dim0_stride[31:0]
  g1[6] = 0;
  g1[7] = 0;
  v4i z4 = {0, 0, 0, 0};                        // 2-D: groups 2/3 unused
  v8i z8 = {0, 0, 0, 0, 0, 0, 0, 0};
  __builtin_amdgcn_tensor_load_to_lds(g0, g1, z4, z4, z8, 0);
}
#endif

// ---- kernel 0: f32 -> f16 weight convert ---------------------------------
__global__ void cvt_f32_to_f16(const float* __restrict__ src,
                               _Float16* __restrict__ dst, int n) {
  int i = blockIdx.x * blockDim.x + threadIdx.x;
  int stride = gridDim.x * blockDim.x;
  for (; i < n; i += stride) dst[i] = (_Float16)src[i];
}

// ---- kernel 1: QKV projection (unchanged; lowering was already good) ------
__global__ void __launch_bounds__(256) qkv_project(
    const float* __restrict__ x,
    const _Float16* __restrict__ Wqh, const float* __restrict__ bq,
    const _Float16* __restrict__ Wkh, const float* __restrict__ bk,
    const _Float16* __restrict__ Wvh, const float* __restrict__ bv,
    _Float16* __restrict__ Qh, _Float16* __restrict__ Kh,
    _Float16* __restrict__ Vt) {
  const int wave = threadIdx.x >> 5;
  const int tt   = blockIdx.x * 8 + wave;   // token tile [0,2048)
  const int lane = threadIdx.x & 31;
  const int m    = lane & 15;
  const int g    = lane >> 4;
  const int row0 = tt * 16;
  const int b    = tt >> 8;
  const int tloc = (tt & 255) * 16;

  const float* xrow = x + (size_t)(row0 + m) * D_MODEL;

  v16h XA[8], XB[8];
#pragma unroll
  for (int dc = 0; dc < 8; ++dc) {
    XA[dc] = cvt_fragA_f32(xrow + dc * 32 + g * 8);
    XB[dc] = cvt_fragB_f32(xrow + dc * 32 + g * 8);
  }

  for (int et = 0; et < 16; ++et) {
    v8f aq = {}, ak = {}, av = {};
#pragma unroll
    for (int dc = 0; dc < 8; ++dc) {
      const _Float16* wq = Wqh + (size_t)(et * 16 + m) * D_MODEL + dc * 32 + g * 8;
      const _Float16* wk = Wkh + (size_t)(et * 16 + m) * D_MODEL + dc * 32 + g * 8;
      const _Float16* wv = Wvh + (size_t)(et * 16 + m) * D_MODEL + dc * 32 + g * 8;
      aq = wmma_f16(load_fragA_f16(wq), XB[dc], aq);   // Q^T tile
      ak = wmma_f16(load_fragA_f16(wk), XB[dc], ak);   // K^T tile
      av = wmma_f16(XA[dc], load_fragB_f16(wv), av);   // V tile
    }
    v8f bqv = *(const v8f*)(bq + et * 16 + g * 8);
    v8f bkv = *(const v8f*)(bk + et * 16 + g * 8);
    float bvv = bv[et * 16 + m];
    v8h hq, hk, hv;
#pragma unroll
    for (int r = 0; r < 8; ++r) {
      hq[r] = (_Float16)((aq[r] + bqv[r]) * 0.0625f);  // fold 1/sqrt(256)
      hk[r] = (_Float16)(ak[r] + bkv[r]);
      hv[r] = (_Float16)(av[r] + bvv);
    }
    *(v8h*)(Qh + (size_t)(row0 + m) * D_MODEL + et * 16 + g * 8) = hq;
    *(v8h*)(Kh + (size_t)(row0 + m) * D_MODEL + et * 16 + g * 8) = hk;
    *(v8h*)(Vt + ((size_t)b * D_MODEL + et * 16 + m) * NTOK + tloc + g * 8) = hv;
  }
}

// ---- kernel 2: flash attention w/ LDS-staged K/V -------------------------
// 8 waves/block share each 32-key K/V chunk through LDS (8x less L2 traffic).
// TDM (tensor_load_to_lds) stages the tiles with hardware padding; fallback
// is a cooperative global->VGPR->LDS copy. Double-buffered.
__global__ void __launch_bounds__(256) attention(
    const _Float16* __restrict__ Qh, const _Float16* __restrict__ Kh,
    const _Float16* __restrict__ Vt, float* __restrict__ out) {
  __shared__ __align__(16) _Float16 Ks[2][32 * K_LDS_STRIDE];
  __shared__ __align__(16) _Float16 Vs[2][256 * V_LDS_STRIDE];

  const int tid  = threadIdx.x;
  const int wave = tid >> 5;
  const int qt   = blockIdx.x * 8 + wave;   // [0,2048), block stays in batch
  const int b    = qt >> 8;
  const int q0   = (qt & 255) * 16;
  const int lane = tid & 31;
  const int m    = lane & 15;
  const int g    = lane >> 4;

  const _Float16* Qb = Qh + (size_t)b * NTOK * D_MODEL;
  const _Float16* Kb = Kh + (size_t)b * NTOK * D_MODEL;
  const _Float16* Vb = Vt + (size_t)b * D_MODEL * NTOK;

  v16h Qf[8];                               // loop-invariant Q B-fragments
#pragma unroll
  for (int dc = 0; dc < 8; ++dc)
    Qf[dc] = load_fragB_f16(Qb + (size_t)(q0 + m) * D_MODEL + dc * 32 + g * 8);

  // ---- chunk staging ------------------------------------------------------
  auto stage = [&](int buf, int k0) {
#if USE_TDM
    if (wave == 0) {
      // K chunk: 32 rows x 256 halves, pad 4 dw per 128 dw -> stride 264
      tdm_load_tile_f16(Kb + (size_t)k0 * D_MODEL,
                        (unsigned)(uintptr_t)&Ks[buf][0],
                        D_MODEL, NTOK, D_MODEL, D_MODEL, 32, 6u, 3u);
      // V chunk: 256 rows x 32 halves, pad 4 dw per 16 dw -> stride 40
      tdm_load_tile_f16(Vb + k0, (unsigned)(uintptr_t)&Vs[buf][0],
                        NTOK, D_MODEL, NTOK, 32, D_MODEL, 3u, 3u);
    }
#else
    {  // cooperative copy: K row = tid/8, 64B seg; V row = tid, 64B
      const _Float16* gk = Kb + (size_t)(k0 + (tid >> 3)) * D_MODEL + (tid & 7) * 32;
      _Float16* lk = &Ks[buf][(tid >> 3) * K_LDS_STRIDE + (tid & 7) * 32];
      v8h k0v = ((const v8h*)gk)[0], k1v = ((const v8h*)gk)[1];
      v8h k2v = ((const v8h*)gk)[2], k3v = ((const v8h*)gk)[3];
      ((v8h*)lk)[0] = k0v; ((v8h*)lk)[1] = k1v;
      ((v8h*)lk)[2] = k2v; ((v8h*)lk)[3] = k3v;
      const _Float16* gv = Vb + (size_t)tid * NTOK + k0;
      _Float16* lv = &Vs[buf][tid * V_LDS_STRIDE];
      v8h v0v = ((const v8h*)gv)[0], v1v = ((const v8h*)gv)[1];
      v8h v2v = ((const v8h*)gv)[2], v3v = ((const v8h*)gv)[3];
      ((v8h*)lv)[0] = v0v; ((v8h*)lv)[1] = v1v;
      ((v8h*)lv)[2] = v2v; ((v8h*)lv)[3] = v3v;
    }
#endif
  };
  auto stage_finish = [&]() {
#if USE_TDM
    if (wave == 0) __builtin_amdgcn_s_wait_tensorcnt(0);
#endif
    __syncthreads();
  };

  v8f acc[16];
#pragma unroll
  for (int nt = 0; nt < 16; ++nt) acc[nt] = (v8f){};
  float mrun = -3.0e38f, lrun = 0.f;

  stage(0, 0);
  stage_finish();

  for (int kc = 0; kc < 128; ++kc) {        // 32 keys per chunk
    const int cur = kc & 1;
    if (kc + 1 < 128) stage(cur ^ 1, (kc + 1) * 32);  // overlap w/ compute

    const _Float16* Kc = &Ks[cur][0];
    const _Float16* Vc = &Vs[cur][0];

    v8f s0 = {}, s1 = {};
#pragma unroll
    for (int dc = 0; dc < 8; ++dc) {
      v16h kf0 = load_fragA_f16(Kc + (m)      * K_LDS_STRIDE + dc * 32 + g * 8);
      v16h kf1 = load_fragA_f16(Kc + (16 + m) * K_LDS_STRIDE + dc * 32 + g * 8);
      s0 = wmma_f16(kf0, Qf[dc], s0);       // S^T rows k..k+15
      s1 = wmma_f16(kf1, Qf[dc], s1);       // S^T rows k+16..k+31
    }
    // online softmax; lane owns q = lane%16; regs r -> k = k0+r+g*8 (+16)
    float lm = s0[0];
#pragma unroll
    for (int r = 0; r < 8; ++r) { lm = fmaxf(lm, s0[r]); lm = fmaxf(lm, s1[r]); }
    lm = xor16_max(lm);
    float mnew  = fmaxf(mrun, lm);
    float alpha = __expf(mrun - mnew);
    v16h pf;
    float ls = 0.f;
#pragma unroll
    for (int r = 0; r < 8; ++r) {
      float p0 = __expf(s0[r] - mnew);
      float p1 = __expf(s1[r] - mnew);
      ls += p0 + p1;
      pf[r]     = (_Float16)p0;             // A-operand halves 0..7
      pf[8 + r] = (_Float16)p1;             // A-operand halves 8..15
    }
    ls   = xor16_add(ls);
    lrun = lrun * alpha + ls;
    mrun = mnew;

    float ar[8];
#pragma unroll
    for (int r = 0; r < 8; ++r) ar[r] = lane_bcast(alpha, r + g * 8);

#pragma unroll
    for (int nt = 0; nt < 16; ++nt) {
#pragma unroll
      for (int r = 0; r < 8; ++r) acc[nt][r] *= ar[r];
      v16h vf = load_fragB_f16(Vc + (nt * 16 + m) * V_LDS_STRIDE + g * 8);
      acc[nt] = wmma_f16(pf, vf, acc[nt]);  // out[q,d] += P * V
    }

    if (kc + 1 < 128) stage_finish();
  }

  float li[8];
#pragma unroll
  for (int r = 0; r < 8; ++r) li[r] = 1.f / lane_bcast(lrun, r + g * 8);

  float* ob = out + (size_t)b * NTOK * D_MODEL;
#pragma unroll
  for (int nt = 0; nt < 16; ++nt)
#pragma unroll
    for (int r = 0; r < 8; ++r)
      ob[(size_t)(q0 + r + g * 8) * D_MODEL + nt * 16 + m] = acc[nt][r] * li[r];
}

// ---- launcher -------------------------------------------------------------
extern "C" void kernel_launch(void* const* d_in, const int* in_sizes, int n_in,
                              void* d_out, int out_size, void* d_ws,
                              size_t ws_size, hipStream_t stream) {
  const float* x  = (const float*)d_in[0];
  const float* Wq = (const float*)d_in[1];
  const float* bq = (const float*)d_in[2];
  const float* Wk = (const float*)d_in[3];
  const float* bk = (const float*)d_in[4];
  const float* Wv = (const float*)d_in[5];
  const float* bv = (const float*)d_in[6];
  float* out = (float*)d_out;

  _Float16* Qh  = (_Float16*)d_ws;
  _Float16* Kh  = Qh + (size_t)ROWS * D_MODEL;
  _Float16* Vt  = Kh + (size_t)ROWS * D_MODEL;
  _Float16* Wqh = Vt + (size_t)ROWS * D_MODEL;
  _Float16* Wkh = Wqh + D_MODEL * D_MODEL;
  _Float16* Wvh = Wkh + D_MODEL * D_MODEL;

  cvt_f32_to_f16<<<64, 256, 0, stream>>>(Wq, Wqh, D_MODEL * D_MODEL);
  cvt_f32_to_f16<<<64, 256, 0, stream>>>(Wk, Wkh, D_MODEL * D_MODEL);
  cvt_f32_to_f16<<<64, 256, 0, stream>>>(Wv, Wvh, D_MODEL * D_MODEL);

  qkv_project<<<256, 256, 0, stream>>>(x, Wqh, bq, Wkh, bk, Wvh, bv, Qh, Kh, Vt);
  attention<<<256, 256, 0, stream>>>(Qh, Kh, Vt, out);
}
